// PositionEncoder_10763188044214
// MI455X (gfx1250) — matrix operand order repaired and checked by reading, not compiled, not run on hardware
//
#include <hip/hip_runtime.h>
#include <cmath>

typedef __attribute__((ext_vector_type(16))) _Float16 v16h;
typedef __attribute__((ext_vector_type(8)))  _Float16 v8h;
typedef __attribute__((ext_vector_type(8)))  float    v8f;

#define B_TOTAL   16384
#define IN_DIM    4
#define PE_DIMS   256
#define OUT_DIM   4096
#define TILE_M    16
#define PE_STRIDE 264     // 256 + 8 halves of padding
#define Y_STRIDE  4100    // 4096 + 4 floats of padding (half-wave bank shift = 32)
#define LN_EPS    1e-5f

// ---------------------------------------------------------------------------
// Kernel 0: W (f32, [4096,256] row-major) -> f16 copy in workspace.
// 2 MB result stays resident in L2 (192 MB) for the entire GEMM.
// ---------------------------------------------------------------------------
__global__ void w_to_f16_kernel(const float* __restrict__ W,
                                _Float16* __restrict__ Wh, int n) {
    int i = blockIdx.x * blockDim.x + threadIdx.x;
    if (i < n) Wh[i] = (_Float16)W[i];
}

// ---------------------------------------------------------------------------
// Kernel 1: fused  PE -> GEMM (WMMA f16) -> bias -> LayerNorm.
// One block = 16 batch rows. 256 threads = 8 wave32s.
// LDS: f16 PE tile [16][264] + f32 y panel [16][4100]  (~271 KB / 320 KB WGP).
// Each wave computes 2 independent 16x16 output tiles per iteration so the
// XDL pipe has two parallel WMMA accumulation chains in flight.
// ---------------------------------------------------------------------------
__global__ __launch_bounds__(256, 1)
void pe_gemm_ln_kernel(const float* __restrict__ state,
                       const _Float16* __restrict__ Wh,
                       const float* __restrict__ bias,
                       const float* __restrict__ gamma,
                       const float* __restrict__ beta,
                       float* __restrict__ out)
{
    extern __shared__ char smem_raw[];
    _Float16* pe_lds = (_Float16*)smem_raw;                                        // [16][PE_STRIDE]
    float*    y_lds  = (float*)(smem_raw + TILE_M * PE_STRIDE * sizeof(_Float16)); // [16][Y_STRIDE]

    const int tid = threadIdx.x;
    const int b0  = blockIdx.x * TILE_M;

    // ---------------- Phase A: sinusoidal PE into LDS (f16) ----------------
    {
        const int m     = tid >> 4;          // batch row within tile (0..15)
        const int kbase = (tid & 15) * 16;   // 16 contiguous PE dims
        float s[IN_DIM];
        #pragma unroll
        for (int c = 0; c < IN_DIM; ++c) {
            float v = state[(size_t)(b0 + m) * IN_DIM + c] * 0.1f; // /MAX_COORD
            s[c] = fminf(fmaxf(v, 0.0f), 1.0f);
        }
        #pragma unroll
        for (int i = 0; i < 16; ++i) {
            int k = kbase + i;
            int c = k >> 6;                  // which coordinate
            int j = k & 63;                  // dim within coordinate
            // freq = 10000^(-2j/64) = exp(-j * ln(10000)/32)
            float freq  = __expf(-(float)j * 0.2878231366242557f);
            float angle = s[c] * freq;
            float v = (j & 1) ? __cosf(angle) : __sinf(angle);
            pe_lds[m * PE_STRIDE + k] = (_Float16)v;
        }
    }
    __syncthreads();

    const int lane = tid & 31;
    const int wave = tid >> 5;
    const int h    = lane >> 4;      // lane half (ISA 16-bit fragment layouts)
    const int ln16 = lane & 15;      // M for A-frag, N for B-frag, N for C/D

    // -------- Phase B: build A fragments once (K = 256 = 8 chunks of 32) ----
    // A 16x32 k order used here (contiguous): lane(h,m), element e -> K = 16h + e
    v16h afrag[8];
    #pragma unroll
    for (int c = 0; c < 8; ++c) {
        const _Float16* base = &pe_lds[ln16 * PE_STRIDE + c * 32 + h * 16];
        union { v16h v; v8h p[2]; } u;
        u.p[0] = *(const v8h*)(base);        // K = 32c + 16h .. +7
        u.p[1] = *(const v8h*)(base + 8);    // K = 32c + 16h + 8 .. +7
        afrag[c] = u.v;
    }

    // Wave w handles tile pairs (w + 16i, w + 8 + 16i), i = 0..15 -> 32 tiles.
    for (int t = wave; t < OUT_DIM / 16; t += 16) {
        const int o0 = t * 16;
        const int o1 = (t + 8) * 16;
        // B 32x16 k order used here (contiguous, same as A): lane(h,n), element e -> K = 16h + e.
        // B[k,n] = W[o0+n, k]  -> 32 contiguous f16 from W row (o0+n).
        const _Float16* wrow0 = Wh + (size_t)(o0 + ln16) * PE_DIMS + h * 16;
        const _Float16* wrow1 = Wh + (size_t)(o1 + ln16) * PE_DIMS + h * 16;
        v8f acc0 = {};
        v8f acc1 = {};
        #pragma unroll
        for (int c = 0; c < 8; ++c) {
            v16h bf0 = *(const v16h*)(wrow0 + c * 32);
            v16h bf1 = *(const v16h*)(wrow1 + c * 32);
            acc0 = __builtin_amdgcn_wmma_f32_16x16x32_f16(
                false, afrag[c], false, bf0, (short)0, acc0, false, false);
            acc1 = __builtin_amdgcn_wmma_f32_16x16x32_f16(
                false, afrag[c], false, bf1, (short)0, acc1, false, false);
        }
        // C/D layout: element v -> (M = v + 8h, N = ln16). Bias depends on N only.
        const float bv0 = bias[o0 + ln16];
        const float bv1 = bias[o1 + ln16];
        #pragma unroll
        for (int v = 0; v < 8; ++v) {
            y_lds[(v + 8 * h) * Y_STRIDE + o0 + ln16] = acc0[v] + bv0;
            y_lds[(v + 8 * h) * Y_STRIDE + o1 + ln16] = acc1[v] + bv1;
        }
    }
    __syncthreads();

    // ---------------- Phase C: LayerNorm, 2 rows per wave -------------------
    #pragma unroll
    for (int rr = 0; rr < 2; ++rr) {
        const int r = wave * 2 + rr;
        const float* yrow = &y_lds[r * Y_STRIDE];
        float sum = 0.0f, sumsq = 0.0f;
        for (int col = lane; col < OUT_DIM; col += 32) {
            float v = yrow[col];
            sum += v;
            sumsq += v * v;
        }
        #pragma unroll
        for (int off = 16; off >= 1; off >>= 1) {      // wave32 reduction
            sum   += __shfl_xor(sum, off, 32);
            sumsq += __shfl_xor(sumsq, off, 32);
        }
        const float mu   = sum * (1.0f / OUT_DIM);
        const float var  = sumsq * (1.0f / OUT_DIM) - mu * mu;
        const float rstd = __frsqrt_rn(var + LN_EPS);
        float* orow = out + (size_t)(b0 + r) * OUT_DIM;
        for (int col = lane; col < OUT_DIM; col += 32) {
            // Output is write-once: non-temporal store keeps W/bias/gamma/beta
            // resident in L2 instead of being evicted by 256 MB of results.
            float o = (yrow[col] - mu) * rstd * gamma[col] + beta[col];
            __builtin_nontemporal_store(o, &orow[col]);
        }
    }
}

// ---------------------------------------------------------------------------
extern "C" void kernel_launch(void* const* d_in, const int* in_sizes, int n_in,
                              void* d_out, int out_size, void* d_ws, size_t ws_size,
                              hipStream_t stream) {
    const float* state = (const float*)d_in[0];   // [16384, 4]
    const float* W     = (const float*)d_in[1];   // [4096, 256]
    const float* bias  = (const float*)d_in[2];   // [4096]
    const float* gamma = (const float*)d_in[3];   // [4096]
    const float* beta  = (const float*)d_in[4];   // [4096]
    float* out = (float*)d_out;                   // [16384, 4096]

    _Float16* Wh = (_Float16*)d_ws;               // 4096*256*2 = 2 MB scratch

    const int wn = OUT_DIM * PE_DIMS;
    w_to_f16_kernel<<<(wn + 255) / 256, 256, 0, stream>>>(W, Wh, wn);

    const size_t smem = (size_t)TILE_M * PE_STRIDE * sizeof(_Float16)
                      + (size_t)TILE_M * Y_STRIDE  * sizeof(float);   // ~271 KB
    pe_gemm_ln_kernel<<<B_TOTAL / TILE_M, 256, smem, stream>>>(
        state, Wh, bias, gamma, beta, out);
}
